// MSDeformAttn_11012296147192
// MI455X (gfx1250) — compile-verified, matrix operand order checked by reading, not run on hardware
//
#include <hip/hip_runtime.h>
#include <hip/hip_bf16.h>
#include <math.h>

// ---------------------------------------------------------------------------
// Types for CDNA5 WMMA (wave32): v_wmma_f32_16x16x32_bf16
// ---------------------------------------------------------------------------
typedef __bf16 bf16_t;
typedef __bf16 v16bf __attribute__((ext_vector_type(16)));
typedef __bf16 v8bf  __attribute__((ext_vector_type(8)));
typedef float  v8f   __attribute__((ext_vector_type(8)));

// Problem geometry (static in the reference)
#define NB      4
#define LQ      1000
#define DMODEL  256
#define NHEADS  8
#define NLVL    4
#define NPTS    4
#define DHEAD   64
#define HD      (NHEADS * DHEAD)        // 512
#define LEN_IN  11253
#define MV      (NB * LEN_IN)           // 45012 rows of value GEMM
#define MV_PAD  45056                   // pad to multiple of 128
#define MQ      (NB * LQ)               // 4000 query rows
#define MQ_PAD  4096                    // pad to multiple of 128

// ---------------------------------------------------------------------------
// f32 -> bf16 conversion with zero tail padding
// ---------------------------------------------------------------------------
__global__ void cvt_f32_bf16(const float* __restrict__ src, bf16_t* __restrict__ dst,
                             long n_valid, long n_total) {
    long i = (long)blockIdx.x * blockDim.x + threadIdx.x;
    if (i < n_total) {
        float v = (i < n_valid) ? src[i] : 0.0f;
        dst[i] = (bf16_t)v;
    }
}

// ---------------------------------------------------------------------------
// Fragment load: 16 bf16 per lane laid out for v_wmma_f32_16x16x32_bf16.
// elements 0..7  = p[0..7]   (K block base)
// elements 8..15 = p[16..23] (K block base+16)
// p is already lane-shifted by (lane>=16 ? 8 : 0). Works for global and LDS.
// ---------------------------------------------------------------------------
__device__ __forceinline__ v16bf load_frag16(const bf16_t* p) {
    v8bf lo = *reinterpret_cast<const v8bf*>(p);
    v8bf hi = *reinterpret_cast<const v8bf*>(p + 16);
    v16bf r;
#pragma unroll
    for (int i = 0; i < 8; ++i) { r[i] = lo[i]; r[i + 8] = hi[i]; }
    return r;
}

// ---------------------------------------------------------------------------
// WMMA GEMM: C[M, Nout] = A[M, K](bf16) * Bt[Nout, K](bf16)^T + bias
// Block = 256 threads = 8 waves; each wave: 16(M) x 64(N); block tile 128 x 64.
// B tile (64 x K) staged once into LDS (padded stride, ds_load_b128 reads).
// Per k-step: next A fragment issued first (global), then all 4 B fragments
// loaded into independent registers (one ds clause), then 4 WMMAs.
// OUT_BF16 selects bf16 vs f32 stores. Stores guarded by Mvalid.
// ---------------------------------------------------------------------------
template <bool OUT_BF16, int K>
__global__ __launch_bounds__(256)
void wmma_gemm_bf16(const bf16_t* __restrict__ A, const bf16_t* __restrict__ Bt,
                    const float* __restrict__ bias, void* __restrict__ Cout,
                    int Mvalid, int Nout) {
    constexpr int BSTRIDE = K + 8;               // +16B pad: de-conflict LDS banks
    __shared__ bf16_t Bs[64 * BSTRIDE];

    const int tid  = threadIdx.x;
    const int lane = tid & 31;
    const int wave = tid >> 5;
    const int m0   = blockIdx.x * 128 + wave * 16;
    const int n0   = blockIdx.y * 64;

    // ---- stage B tile [64 x K] into LDS, coalesced 16-byte chunks ----
    {
        constexpr int CPR = K / 8;               // 16B chunks per row
        constexpr int CHUNKS = 64 * CPR;
#pragma unroll
        for (int c = tid; c < CHUNKS; c += 256) {
            const int r  = c / CPR;
            const int kk = (c - r * CPR) * 8;
            *reinterpret_cast<v8bf*>(&Bs[r * BSTRIDE + kk]) =
                *reinterpret_cast<const v8bf*>(Bt + (size_t)(n0 + r) * K + kk);
        }
    }
    __syncthreads();

    const int l15    = lane & 15;
    const int kshift = (lane >> 4) * 8;          // lanes 16..31: K blocks 8..15 / 24..31

    const bf16_t* aRow = A + (size_t)(m0 + l15) * K + kshift;
    const bf16_t* bRow = &Bs[l15 * BSTRIDE + kshift];

    v8f acc[4];
#pragma unroll
    for (int j = 0; j < 4; ++j)
#pragma unroll
        for (int e = 0; e < 8; ++e) acc[j][e] = 0.0f;

    // ---- K loop: A pipelined from global, B fragments batched from LDS ----
    v16bf afrag = load_frag16(aRow);
#pragma unroll
    for (int k0 = 0; k0 < K; k0 += 32) {
        v16bf a_next = afrag;
        if (k0 + 32 < K) a_next = load_frag16(aRow + k0 + 32);   // covers 4 WMMAs
        v16bf bfr[4];
#pragma unroll
        for (int j = 0; j < 4; ++j)                               // one ds clause
            bfr[j] = load_frag16(bRow + j * 16 * BSTRIDE + k0);
#pragma unroll
        for (int j = 0; j < 4; ++j)
            acc[j] = __builtin_amdgcn_wmma_f32_16x16x32_bf16(
                false, afrag, false, bfr[j], (short)0, acc[j], false, false);
        afrag = a_next;
    }

    // ---- store: lanes 0..15 rows M=0..7, lanes 16..31 rows M=8..15 ----
    const int rbase = (lane < 16) ? 0 : 8;
#pragma unroll
    for (int j = 0; j < 4; ++j) {
        const int col = n0 + j * 16 + l15;
        const float bv = bias ? bias[col] : 0.0f;
#pragma unroll
        for (int r = 0; r < 8; ++r) {
            const int row = m0 + rbase + r;
            if (row < Mvalid) {
                float v = acc[j][r] + bv;
                if (OUT_BF16)
                    ((bf16_t*)Cout)[(size_t)row * Nout + col] = (bf16_t)v;
                else
                    ((float*)Cout)[(size_t)row * Nout + col] = v;
            }
        }
    }
}

// ---------------------------------------------------------------------------
// Softmax over groups of 16 (in place). One thread per (n, q, h).
// ---------------------------------------------------------------------------
__global__ void softmax16_kernel(float* __restrict__ x, int ngroups) {
    int i = blockIdx.x * blockDim.x + threadIdx.x;
    if (i >= ngroups) return;
    float* p = x + (size_t)i * 16;
    float m = p[0];
#pragma unroll
    for (int j = 1; j < 16; ++j) m = fmaxf(m, p[j]);
    float s = 0.0f;
#pragma unroll
    for (int j = 0; j < 16; ++j) { float e = __expf(p[j] - m); p[j] = e; s += e; }
    float inv = 1.0f / s;
#pragma unroll
    for (int j = 0; j < 16; ++j) p[j] *= inv;
}

// ---------------------------------------------------------------------------
// Deformable sampling: grid (N*Lq, 2), block 256 = 4 heads x 64 channels.
// value: bf16 [MV rows][512]; off: f32 [MQ][256]; aw: f32 [MQ][128];
// ref: f32 [N, Lq, L, 2]; out: f32 [MQ][512]
// ---------------------------------------------------------------------------
__global__ __launch_bounds__(256)
void ms_sample_kernel(const bf16_t* __restrict__ value,
                      const float* __restrict__ ref,
                      const float* __restrict__ off,
                      const float* __restrict__ aw,
                      float* __restrict__ out) {
    const int nq = blockIdx.x;                       // n*Lq + q
    const int h  = blockIdx.y * 4 + (threadIdx.x >> 6);
    const int d  = threadIdx.x & 63;
    const int n  = nq / LQ;

    const int hs[NLVL] = {92, 46, 23, 12};
    const int wl[NLVL] = {92, 46, 23, 12};
    const int st[NLVL] = {0, 8464, 10580, 11109};

    const float* offq = off + (size_t)nq * (NHEADS * NLVL * NPTS * 2);
    const float* awq  = aw  + (size_t)nq * (NHEADS * NLVL * NPTS);
    const size_t vbase = (size_t)n * LEN_IN;
    const int ch = h * DHEAD + d;

    float acc = 0.0f;
#pragma unroll
    for (int l = 0; l < NLVL; ++l) {
        const int H = hs[l], W = wl[l];
        const float rx = ref[((size_t)nq * NLVL + l) * 2 + 0];
        const float ry = ref[((size_t)nq * NLVL + l) * 2 + 1];
        const size_t lvbase = (vbase + st[l]) * (size_t)HD;
#pragma unroll
        for (int p = 0; p < NPTS; ++p) {
            const int pi = (h * NLVL + l) * NPTS + p;
            const float ox = offq[pi * 2 + 0];
            const float oy = offq[pi * 2 + 1];
            const float wt = awq[pi];
            // loc = ref + off/(w,h); grid-sample coords: loc*size - 0.5
            const float x = rx * (float)W + ox - 0.5f;
            const float y = ry * (float)H + oy - 0.5f;
            const float xf = floorf(x), yf = floorf(y);
            const int x0 = (int)xf, y0 = (int)yf;
            const float lx = x - xf, ly = y - yf;
            const float w00 = (1.0f - lx) * (1.0f - ly);
            const float w01 = lx * (1.0f - ly);
            const float w10 = (1.0f - lx) * ly;
            const float w11 = lx * ly;
            float s = 0.0f;
#pragma unroll
            for (int cy = 0; cy < 2; ++cy) {
#pragma unroll
                for (int cx = 0; cx < 2; ++cx) {
                    const int yy = y0 + cy, xx = x0 + cx;
                    if (xx >= 0 && xx < W && yy >= 0 && yy < H) {
                        const float cw = (cy == 0) ? (cx == 0 ? w00 : w01)
                                                   : (cx == 0 ? w10 : w11);
                        s += cw * (float)value[lvbase + (size_t)(yy * W + xx) * HD + ch];
                    }
                }
            }
            acc += wt * s;
        }
    }
    out[(size_t)nq * HD + ch] = acc;
}

// ---------------------------------------------------------------------------
// Launch
// ---------------------------------------------------------------------------
static inline size_t align256(size_t x) { return (x + 255) & ~(size_t)255; }

extern "C" void kernel_launch(void* const* d_in, const int* in_sizes, int n_in,
                              void* d_out, int out_size, void* d_ws, size_t ws_size,
                              hipStream_t stream) {
    const float* q_f32   = (const float*)d_in[0];   // [4,1000,256]
    const float* refpts  = (const float*)d_in[1];   // [4,1000,4,2]
    const float* in_flat = (const float*)d_in[2];   // [4,11253,256]
    const float* W_off   = (const float*)d_in[5];   // [256,256]
    const float* b_off   = (const float*)d_in[6];   // [256]
    const float* W_attn  = (const float*)d_in[7];   // [128,256]
    const float* b_attn  = (const float*)d_in[8];   // [128]
    const float* W_val   = (const float*)d_in[9];   // [512,256]
    const float* b_val   = (const float*)d_in[10];  // [512]
    const float* W_out   = (const float*)d_in[11];  // [256,512]
    const float* b_out   = (const float*)d_in[12];  // [256]
    float* out_f32 = (float*)d_out;                 // [4,1000,256]

    // Workspace layout
    char* ws = (char*)d_ws;
    size_t o = 0;
    bf16_t* A_bf      = (bf16_t*)(ws + o); o = align256(o + (size_t)MV_PAD * DMODEL * 2);
    bf16_t* value_bf  = (bf16_t*)(ws + o); o = align256(o + (size_t)MV_PAD * HD * 2);
    bf16_t* q_bf      = (bf16_t*)(ws + o); o = align256(o + (size_t)MQ_PAD * DMODEL * 2);
    bf16_t* wval_bf   = (bf16_t*)(ws + o); o = align256(o + (size_t)HD * DMODEL * 2);
    bf16_t* woff_bf   = (bf16_t*)(ws + o); o = align256(o + (size_t)256 * DMODEL * 2);
    bf16_t* wattn_bf  = (bf16_t*)(ws + o); o = align256(o + (size_t)128 * DMODEL * 2);
    bf16_t* wout_bf   = (bf16_t*)(ws + o); o = align256(o + (size_t)DMODEL * HD * 2);
    float*  off_raw   = (float*)(ws + o);  o = align256(o + (size_t)MQ_PAD * 256 * 4);
    float*  attn_raw  = (float*)(ws + o);  o = align256(o + (size_t)MQ_PAD * 128 * 4);
    float*  attout    = (float*)(ws + o);  o = align256(o + (size_t)MQ * HD * 4);
    bf16_t* attout_bf = (bf16_t*)(ws + o); o = align256(o + (size_t)MQ_PAD * HD * 2);

    const int CT = 256;
    auto nblk = [](long n, int b) { return (unsigned)((n + b - 1) / b); };

    // 1) bf16 conversions (zero-padded tails)
    cvt_f32_bf16<<<nblk((long)MV_PAD * DMODEL, CT), CT, 0, stream>>>(
        in_flat, A_bf, (long)MV * DMODEL, (long)MV_PAD * DMODEL);
    cvt_f32_bf16<<<nblk((long)MQ_PAD * DMODEL, CT), CT, 0, stream>>>(
        q_f32, q_bf, (long)MQ * DMODEL, (long)MQ_PAD * DMODEL);
    cvt_f32_bf16<<<nblk((long)HD * DMODEL, CT), CT, 0, stream>>>(
        W_val, wval_bf, (long)HD * DMODEL, (long)HD * DMODEL);
    cvt_f32_bf16<<<nblk((long)256 * DMODEL, CT), CT, 0, stream>>>(
        W_off, woff_bf, (long)256 * DMODEL, (long)256 * DMODEL);
    cvt_f32_bf16<<<nblk((long)128 * DMODEL, CT), CT, 0, stream>>>(
        W_attn, wattn_bf, (long)128 * DMODEL, (long)128 * DMODEL);
    cvt_f32_bf16<<<nblk((long)DMODEL * HD, CT), CT, 0, stream>>>(
        W_out, wout_bf, (long)DMODEL * HD, (long)DMODEL * HD);

    // 2) value = input_flatten @ W_val^T + b_val  -> bf16 [MV_PAD, 512]
    wmma_gemm_bf16<true, DMODEL><<<dim3(MV_PAD / 128, HD / 64), 256, 0, stream>>>(
        A_bf, wval_bf, b_val, value_bf, MV_PAD, HD);

    // 3) offsets = query @ W_off^T + b_off -> f32 [MQ_PAD, 256]
    wmma_gemm_bf16<false, DMODEL><<<dim3(MQ_PAD / 128, 256 / 64), 256, 0, stream>>>(
        q_bf, woff_bf, b_off, off_raw, MQ_PAD, 256);

    // 4) attn logits = query @ W_attn^T + b_attn -> f32 [MQ_PAD, 128]
    wmma_gemm_bf16<false, DMODEL><<<dim3(MQ_PAD / 128, 128 / 64), 256, 0, stream>>>(
        q_bf, wattn_bf, b_attn, attn_raw, MQ_PAD, 128);

    // 5) softmax over 16 (L*P) per (n,q,h)
    softmax16_kernel<<<nblk((long)MQ * NHEADS, CT), CT, 0, stream>>>(
        attn_raw, MQ * NHEADS);

    // 6) deformable bilinear sampling -> f32 [MQ, 512]
    ms_sample_kernel<<<dim3(MQ, NHEADS / 4), 256, 0, stream>>>(
        value_bf, refpts, off_raw, attn_raw, attout);

    // 7) bf16 conversion of attention output (pad rows to 4096)
    cvt_f32_bf16<<<nblk((long)MQ_PAD * HD, CT), CT, 0, stream>>>(
        attout, attout_bf, (long)MQ * HD, (long)MQ_PAD * HD);

    // 8) final = attout @ W_out^T + b_out -> d_out [4000, 256] (guarded, K=512)
    wmma_gemm_bf16<false, HD><<<dim3(MQ_PAD / 128, DMODEL / 64), 256, 0, stream>>>(
        attout_bf, wout_bf, b_out, out_f32, MQ, DMODEL);
}
